// QuantizedLinearLayer_89060441849929
// MI455X (gfx1250) — matrix operand order repaired and checked by reading, not compiled, not run on hardware
//
#include <hip/hip_runtime.h>

typedef __attribute__((ext_vector_type(16))) _Float16 v16h;
typedef __attribute__((ext_vector_type(8)))  _Float16 v8h;
typedef __attribute__((ext_vector_type(4)))  _Float16 v4h;
typedef __attribute__((ext_vector_type(8)))  float    v8f;
typedef __attribute__((ext_vector_type(4)))  float    f4v;
typedef __attribute__((ext_vector_type(4)))  int      i4v;

#define DIN   4096
#define DOUT  4096
#define BM    128
#define BN    256
#define KT    32
#define LDSTR 40   // 32 halves + 8 pad -> 80B row stride, bank-skewed

union FragU { v16h v; v8h h[2]; };

__global__ __launch_bounds__(256, 1)
void qlinear_wmma_kernel(const float* __restrict__ X,
                         const int*   __restrict__ Wq,
                         const float* __restrict__ scales,
                         const float* __restrict__ bias,
                         float*       __restrict__ Out)
{
    __shared__ _Float16 As[2][BM * LDSTR];   // 128 rows  x 32 K (f16)
    __shared__ _Float16 Bs[2][BN * LDSTR];   // 256 rows  x 32 K (f16)

    const int tid   = threadIdx.x;
    const int lane  = tid & 31;
    const int l16   = lane & 15;
    const int lhalf = lane >> 4;          // 0 or 1
    const int wave  = tid >> 5;           // 0..7
    const int wm    = wave & 1;           // wave m-offset = wm*64
    const int wn    = wave >> 1;          // wave n-offset = wn*64  (0..3)

    const int mTile = blockIdx.y * BM;
    const int nTile = blockIdx.x * BN;

    const float* Xt = X  + (size_t)mTile * DIN;
    const int*   Wt = Wq + (size_t)nTile * DIN;

    // per-stage staging registers: 4 x float4 (A) + 8 x int4 (B) per thread
    f4v aS[4];
    i4v bS[8];

    v8f acc[4][4];
    v8f zero = {};
#pragma unroll
    for (int i = 0; i < 4; ++i)
#pragma unroll
        for (int j = 0; j < 4; ++j)
            acc[i][j] = zero;

    auto gload = [&](int kt) {
#pragma unroll
        for (int i = 0; i < 4; ++i) {                 // A: 128 rows x 8 float4
            int u   = tid + (i << 8);
            int row = u >> 3;
            int c   = (u & 7) << 2;
            aS[i] = *(const f4v*)(Xt + (size_t)row * DIN + kt + c);
        }
#pragma unroll
        for (int i = 0; i < 8; ++i) {                 // B: 256 rows x 8 int4
            int u   = tid + (i << 8);
            int row = u >> 3;
            int c   = (u & 7) << 2;
            bS[i] = *(const i4v*)(Wt + (size_t)row * DIN + kt + c);
        }
    };

    auto lstore = [&](int buf) {
#pragma unroll
        for (int i = 0; i < 4; ++i) {
            int u   = tid + (i << 8);
            int row = u >> 3;
            int c   = (u & 7) << 2;
            f4v a = aS[i];
            v4h ha;
            ha.x = (_Float16)a.x; ha.y = (_Float16)a.y;
            ha.z = (_Float16)a.z; ha.w = (_Float16)a.w;
            *(v4h*)&As[buf][row * LDSTR + c] = ha;        // ds_store_b64
        }
#pragma unroll
        for (int i = 0; i < 8; ++i) {
            int u   = tid + (i << 8);
            int row = u >> 3;
            int c   = (u & 7) << 2;
            i4v b = bS[i];
            v4h hb;
            hb.x = (_Float16)b.x; hb.y = (_Float16)b.y;   // int8 values: exact in f16
            hb.z = (_Float16)b.z; hb.w = (_Float16)b.w;
            *(v4h*)&Bs[buf][row * LDSTR + c] = hb;
        }
    };

    auto compute = [&](int buf) {
        // B fragments (32x16 f16): n = l16; lanes 0-15 hold K 0..15, lanes 16-31 K 16..31
        v16h bF[4];
#pragma unroll
        for (int j = 0; j < 4; ++j) {
            int row = wn * 64 + j * 16 + l16;
            const _Float16* p = &Bs[buf][row * LDSTR + lhalf * 16];
            FragU f;
            f.h[0] = *(const v8h*)(p);         // ds_load_b128
            f.h[1] = *(const v8h*)(p + 8);     // ds_load_b128 (contiguous 32B)
            bF[j] = f.v;
        }
        // A fragment (16x32 f16): lane l16 = row; lanes 0-15 hold K 0..15,
        // lanes 16-31 hold K 16..31
#pragma unroll
        for (int i = 0; i < 4; ++i) {
            int row = wm * 64 + i * 16 + l16;
            const _Float16* p = &As[buf][row * LDSTR + lhalf * 16];
            FragU f;
            f.h[0] = *(const v8h*)(p);
            f.h[1] = *(const v8h*)(p + 8);
            v16h aF = f.v;
#pragma unroll
            for (int j = 0; j < 4; ++j)
                acc[i][j] = __builtin_amdgcn_wmma_f32_16x16x32_f16(
                    false, aF, false, bF[j], (short)0, acc[i][j], false, false);
        }
    };

    // ---- pipelined main loop over K ----
    gload(0);
    lstore(0);
    __syncthreads();

    int buf = 0;
    for (int kt = 0; kt < DIN; kt += KT) {
        const bool more = (kt + KT) < DIN;
        if (more) gload(kt + KT);      // issue next-stage loads
        compute(buf);                  // hide load latency under 16 WMMAs
        if (more) lstore(buf ^ 1);     // convert + stage into other buffer
        __syncthreads();
        buf ^= 1;
    }

    // ---- epilogue: out = acc * scale[n] + bias[n] ----
    float sc[4], bi[4];
#pragma unroll
    for (int j = 0; j < 4; ++j) {
        int n = nTile + wn * 64 + j * 16 + l16;
        sc[j] = scales[n];
        bi[j] = bias[n];
    }

#pragma unroll
    for (int i = 0; i < 4; ++i) {
        int mbase = mTile + wm * 64 + i * 16 + lhalf * 8;  // C/D layout: lanes 16-31 -> M+8
#pragma unroll
        for (int j = 0; j < 4; ++j) {
            int n = nTile + wn * 64 + j * 16 + l16;
#pragma unroll
            for (int r = 0; r < 8; ++r) {
                Out[(size_t)(mbase + r) * DOUT + n] = acc[i][j][r] * sc[j] + bi[j];
            }
        }
    }
}

extern "C" void kernel_launch(void* const* d_in, const int* in_sizes, int n_in,
                              void* d_out, int out_size, void* d_ws, size_t ws_size,
                              hipStream_t stream) {
    const float* x  = (const float*)d_in[0];
    const int*   wq = (const int*)d_in[1];
    const float* sc = (const float*)d_in[2];
    const float* bi = (const float*)d_in[3];
    float* out = (float*)d_out;

    const int tokens = in_sizes[0] / DIN;        // 8192
    dim3 grid(DOUT / BN, tokens / BM);           // (16, 64)
    qlinear_wmma_kernel<<<grid, dim3(256, 1, 1), 0, stream>>>(x, wq, sc, bi, out);
}
